// GCN_SAGE_residual_11914239279204
// MI455X (gfx1250) — compile-verified, matrix operand order checked by reading, not run on hardware
//
#include <hip/hip_runtime.h>
#include <hip/hip_bf16.h>
#include <stdint.h>

// ---------------------------------------------------------------------------
// GraphSAGE (mean) x2 + global LayerNorm + ReLU + residual, for gfx1250.
// GEMMs use v_wmma_f32_16x16x32_bf16 (CDNA5 WMMA, wave32) with LDS-staged
// weights (conflict-free chunk-major layout).
// ---------------------------------------------------------------------------

#define GN      100000
#define GE      1600000
#define GC      128          // channel dim (both D and H are 128)
#define GEPS    1e-5f

typedef __attribute__((ext_vector_type(16))) __bf16 v16bf;
typedef __attribute__((ext_vector_type(8)))  float  v8f;

union Frag16 {               // one WMMA A/B operand: 16 bf16 per lane = 8 VGPRs
    v16bf v;
    uint4 u[2];
};

static __device__ __forceinline__ unsigned short f32_to_bf16_rne(float f) {
    unsigned int u = __float_as_uint(f);
    unsigned int r = u + 0x7FFFu + ((u >> 16) & 1u);   // round-to-nearest-even
    return (unsigned short)(r >> 16);
}

// --- weight transpose + f32->bf16:  Wt[n*128+k] = bf16(W[k*128+n]) ----------
__global__ void k_wconv_t(const float* __restrict__ W, unsigned short* __restrict__ Wt) {
    int i = blockIdx.x * blockDim.x + threadIdx.x;   // 128*128 threads
    int k = i >> 7, n = i & 127;
    Wt[n * GC + k] = f32_to_bf16_rne(W[k * GC + n]);
}

// --- plain f32 -> bf16 convert ---------------------------------------------
__global__ void k_convbf(const float* __restrict__ in, unsigned short* __restrict__ out, int n) {
    int i = blockIdx.x * blockDim.x + threadIdx.x;
    if (i < n) out[i] = f32_to_bf16_rne(in[i]);
}

// --- per-destination degree count ------------------------------------------
__global__ void k_degree(const long long* __restrict__ ei, float* __restrict__ cnt) {
    int e = blockIdx.x * blockDim.x + threadIdx.x;
    if (e >= GE) return;
    int dst = (int)ei[GE + e];
    atomicAdd(&cnt[dst], 1.0f);
}

// --- scatter-add of source features: 1 wave per edge, 4 ch per lane --------
__global__ void k_scatter(const float* __restrict__ h, const long long* __restrict__ ei,
                          float* __restrict__ agg) {
    long long gid = (long long)blockIdx.x * blockDim.x + threadIdx.x;
    long long total = (long long)GE * 32;
    if (gid >= total) return;
    int e = (int)(gid >> 5);
    int c = ((int)gid & 31) * 4;
    int src = (int)ei[e];
    int dst = (int)ei[GE + e];
    float4 v = *(const float4*)(h + (size_t)src * GC + c);
    float* a = agg + (size_t)dst * GC + c;
    atomicAdd(a + 0, v.x);
    atomicAdd(a + 1, v.y);
    atomicAdd(a + 2, v.z);
    atomicAdd(a + 3, v.w);
}

// --- agg = agg / max(cnt,1), emit bf16 -------------------------------------
__global__ void k_finalize_agg(const float* __restrict__ agg, const float* __restrict__ cnt,
                               unsigned short* __restrict__ aggbf, int n) {
    int i = blockIdx.x * blockDim.x + threadIdx.x;
    if (i >= n) return;
    float d = fmaxf(cnt[i >> 7], 1.0f);
    aggbf[i] = f32_to_bf16_rne(agg[i] / d);
}

// --- WMMA GEMM:  out[N x 128] = Abf @ Wl + Xbf @ Wr + bias -----------------
// Weights arrive transposed+bf16 (WT[n*128+k]); viewed as uint4 chunks
// WT4[n*16 + cb] where chunk cb covers k = cb*8 .. cb*8+7.
// LDS layout is chunk-major s[cb*128 + n]: lanes 0..15 of a B-fragment read
// 16 consecutive uint4s -> all 64 LDS banks hit exactly once (no conflicts).
__global__ void __launch_bounds__(256)
k_gemm_wmma(const unsigned short* __restrict__ Abf,   // agg, bf16 [N x 128]
            const unsigned short* __restrict__ Xbf,   // root, bf16 [N x 128]
            const uint4* __restrict__ Wl4,            // bf16 WT, uint4 view [128*16]
            const uint4* __restrict__ Wr4,            // bf16 WT, uint4 view [128*16]
            const float* __restrict__ bias,           // [128]
            float* __restrict__ out)                  // f32 [N x 128]
{
    extern __shared__ uint4 smem4[];                  // 2 * 16 * 128 uint4 = 64 KB
    uint4* sWl = smem4;
    uint4* sWr = smem4 + 16 * 128;

    // ---- cooperative stage: coalesced global read, chunk-major LDS write --
    for (int idx = threadIdx.x; idx < 2048; idx += 256) {
        int n = idx >> 4, cb = idx & 15;
        sWl[cb * 128 + n] = Wl4[idx];
        sWr[cb * 128 + n] = Wr4[idx];
    }
    __syncthreads();                                   // before any divergent exit

    int wave = (blockIdx.x * (int)blockDim.x + (int)threadIdx.x) >> 5;
    if (wave * 16 >= GN) return;                       // wave-uniform tail exit
    int lane = threadIdx.x & 31;
    int row0 = wave * 16;
    int rsub = lane & 15;                              // M (for A) / N (for B,C)
    int half = lane >> 4;                              // K sub-half select
    int koff = half * 8;

    // ---- load A fragments for both inputs (4 K-tiles each) ----
    Frag16 Aagg[4], Ax[4];
    const size_t arow = (size_t)(row0 + rsub) * GC;
#pragma unroll
    for (int kt = 0; kt < 4; ++kt) {
        size_t b = arow + kt * 32 + koff;
        Aagg[kt].u[0] = *(const uint4*)(Abf + b);
        Aagg[kt].u[1] = *(const uint4*)(Abf + b + 16);
        Ax[kt].u[0]   = *(const uint4*)(Xbf + b);
        Ax[kt].u[1]   = *(const uint4*)(Xbf + b + 16);
    }

#pragma unroll
    for (int nt = 0; nt < 8; ++nt) {
        int n = nt * 16 + rsub;
        v8f c = {};                                    // zero C -> inline-0 SRC2

#pragma unroll
        for (int kt = 0; kt < 4; ++kt) {
            Frag16 B;
            int cb = kt * 4 + half;
            B.u[0] = sWl[cb * 128 + n];
            B.u[1] = sWl[(cb + 2) * 128 + n];
            c = __builtin_amdgcn_wmma_f32_16x16x32_bf16(
                    false, Aagg[kt].v, false, B.v, (short)0, c, false, false);
        }
#pragma unroll
        for (int kt = 0; kt < 4; ++kt) {
            Frag16 B;
            int cb = kt * 4 + half;
            B.u[0] = sWr[cb * 128 + n];
            B.u[1] = sWr[(cb + 2) * 128 + n];
            c = __builtin_amdgcn_wmma_f32_16x16x32_bf16(
                    false, Ax[kt].v, false, B.v, (short)0, c, false, false);
        }

        // C/D layout: lane = N col, VGPR v = M row (+8 for lanes 16..31)
        float bv = bias[n];
#pragma unroll
        for (int v = 0; v < 8; ++v) {
            int m = row0 + v + half * 8;
            out[(size_t)m * GC + n] = c[v] + bv;
        }
    }
}

// --- global sum / sum-of-squares reduction (float4 streaming) --------------
__global__ void k_reduce(const float* __restrict__ h, int n, float* __restrict__ sums) {
    __shared__ float s1[256], s2[256];
    const float4* h4 = (const float4*)h;
    int n4 = n >> 2;
    float a = 0.f, b = 0.f;
    for (int i = blockIdx.x * blockDim.x + threadIdx.x; i < n4; i += gridDim.x * blockDim.x) {
        float4 v = h4[i];
        a += v.x + v.y + v.z + v.w;
        b += v.x * v.x + v.y * v.y + v.z * v.z + v.w * v.w;
    }
    s1[threadIdx.x] = a; s2[threadIdx.x] = b;
    __syncthreads();
    for (int s = 128; s > 0; s >>= 1) {
        if ((int)threadIdx.x < s) {
            s1[threadIdx.x] += s1[threadIdx.x + s];
            s2[threadIdx.x] += s2[threadIdx.x + s];
        }
        __syncthreads();
    }
    if (threadIdx.x == 0) {
        atomicAdd(&sums[0], s1[0]);
        atomicAdd(&sums[1], s2[0]);
    }
}

// --- layer-1 epilogue: graph-LN + affine + relu, emit f32 + bf16 -----------
__global__ void k_norm_relu_bf(const float* __restrict__ hraw, const float* __restrict__ sums,
                               const float* __restrict__ g, const float* __restrict__ b,
                               float* __restrict__ hout, unsigned short* __restrict__ hbf, int n) {
    int i = blockIdx.x * blockDim.x + threadIdx.x;
    if (i >= n) return;
    float inv = 1.0f / (float)n;
    float mu  = sums[0] * inv;
    float var = sums[1] * inv - mu * mu;
    float rs  = rsqrtf(var + GEPS);
    int c = i & 127;
    float v = (hraw[i] - mu) * rs * g[c] + b[c];
    v = fmaxf(v, 0.0f);
    hout[i] = v;
    hbf[i]  = f32_to_bf16_rne(v);
}

// --- layer-2 epilogue: graph-LN + affine + relu + residual (in-place ok) ---
__global__ void k_norm_relu_add(const float* __restrict__ hraw, const float* __restrict__ sums,
                                const float* __restrict__ g, const float* __restrict__ b,
                                const float* __restrict__ x, float* __restrict__ out, int n) {
    int i = blockIdx.x * blockDim.x + threadIdx.x;
    if (i >= n) return;
    float inv = 1.0f / (float)n;
    float mu  = sums[0] * inv;
    float var = sums[1] * inv - mu * mu;
    float rs  = rsqrtf(var + GEPS);
    int c = i & 127;
    float v = (hraw[i] - mu) * rs * g[c] + b[c];
    v = fmaxf(v, 0.0f);
    out[i] = v + x[i];
}

// ---------------------------------------------------------------------------
extern "C" void kernel_launch(void* const* d_in, const int* in_sizes, int n_in,
                              void* d_out, int out_size, void* d_ws, size_t ws_size,
                              hipStream_t stream) {
    const float*     x   = (const float*)d_in[0];
    const long long* ei  = (const long long*)d_in[1];
    const float*     Wl1 = (const float*)d_in[2];
    const float*     bl1 = (const float*)d_in[3];
    const float*     Wr1 = (const float*)d_in[4];
    const float*     g1  = (const float*)d_in[5];
    const float*     b1  = (const float*)d_in[6];
    const float*     Wl2 = (const float*)d_in[7];
    const float*     bl2 = (const float*)d_in[8];
    const float*     Wr2 = (const float*)d_in[9];
    const float*     g2  = (const float*)d_in[10];
    const float*     b2  = (const float*)d_in[11];
    float* out = (float*)d_out;

    const int NC = GN * GC;                       // 12.8M elements

    // ---- workspace layout (256B aligned slots) ----
    char* base = (char*)d_ws;
    size_t off = 0;
    auto take = [&](size_t bytes) -> char* {
        char* p = base + off;
        off = (off + bytes + 255) & ~(size_t)255;
        return p;
    };
    float*          cnt   = (float*)take((size_t)GN * 4);
    float*          agg   = (float*)take((size_t)NC * 4);
    unsigned short* aggbf = (unsigned short*)take((size_t)NC * 2);
    unsigned short* hbf   = (unsigned short*)take((size_t)NC * 2);  // xbf then h1bf
    float*          h1    = (float*)take((size_t)NC * 4);
    unsigned short* wl1t  = (unsigned short*)take((size_t)GC * GC * 2);
    unsigned short* wr1t  = (unsigned short*)take((size_t)GC * GC * 2);
    unsigned short* wl2t  = (unsigned short*)take((size_t)GC * GC * 2);
    unsigned short* wr2t  = (unsigned short*)take((size_t)GC * GC * 2);
    float*          sums  = (float*)take(4 * 4);                    // [s1,sq1,s2,sq2]

    // ---- zero accumulators ----
    hipMemsetAsync(cnt,  0, (size_t)GN * 4, stream);
    hipMemsetAsync(agg,  0, (size_t)NC * 4, stream);
    hipMemsetAsync(sums, 0, 16, stream);

    const int T = 256;
    const int gEW = 64;                            // 128*128/256 blocks per weight
    k_wconv_t<<<gEW, T, 0, stream>>>(Wl1, wl1t);
    k_wconv_t<<<gEW, T, 0, stream>>>(Wr1, wr1t);
    k_wconv_t<<<gEW, T, 0, stream>>>(Wl2, wl2t);
    k_wconv_t<<<gEW, T, 0, stream>>>(Wr2, wr2t);
    k_convbf<<<(NC + T - 1) / T, T, 0, stream>>>(x, hbf, NC);       // xbf

    k_degree<<<(GE + T - 1) / T, T, 0, stream>>>(ei, cnt);

    // ---- layer 1 ----
    {
        long long st = (long long)GE * 32;
        k_scatter<<<(int)((st + T - 1) / T), T, 0, stream>>>(x, ei, agg);
    }
    k_finalize_agg<<<(NC + T - 1) / T, T, 0, stream>>>(agg, cnt, aggbf, NC);

    const int gGemm = (GN / 16 + 7) / 8;           // 8 waves/block, 16 rows/wave
    const size_t smemB = 2u * 16u * 128u * sizeof(uint4);   // 64 KB
    k_gemm_wmma<<<gGemm, T, smemB, stream>>>(aggbf, hbf,
                                             (const uint4*)wl1t, (const uint4*)wr1t,
                                             bl1, h1);

    k_reduce<<<2048, T, 0, stream>>>(h1, NC, sums);
    k_norm_relu_bf<<<(NC + T - 1) / T, T, 0, stream>>>(h1, sums, g1, b1, h1, hbf, NC);

    // ---- layer 2 ----
    hipMemsetAsync(agg, 0, (size_t)NC * 4, stream);
    {
        long long st = (long long)GE * 32;
        k_scatter<<<(int)((st + T - 1) / T), T, 0, stream>>>(h1, ei, agg);
    }
    k_finalize_agg<<<(NC + T - 1) / T, T, 0, stream>>>(agg, cnt, aggbf, NC);

    k_gemm_wmma<<<gGemm, T, smemB, stream>>>(aggbf, hbf,
                                             (const uint4*)wl2t, (const uint4*)wr2t,
                                             bl2, out);

    k_reduce<<<2048, T, 0, stream>>>(out, NC, sums + 2);
    k_norm_relu_add<<<(NC + T - 1) / T, T, 0, stream>>>(out, sums + 2, g2, b2, x, out, NC);
}
